// TemporalWindowTransformerBlock_54674933678809
// MI455X (gfx1250) — matrix operand order repaired and checked by reading, not compiled, not run on hardware
//
#include <hip/hip_runtime.h>
#include <hip/hip_bf16.h>
#include <math.h>

typedef __attribute__((ext_vector_type(8)))  __bf16 v8bf;
typedef __attribute__((ext_vector_type(16))) __bf16 v16bf;
typedef __attribute__((ext_vector_type(8)))  float  v8f;

#define DIM     512
#define HEADS   8
#define HD      64
#define WIN     10
#define SHIFT   5
#define FRAMES  1000
#define HIDDEN  2048
#define BATCH   64
#define TOKENS  (BATCH * FRAMES)   // 64000
#define LN_EPS  1e-5f

// ---------------------------------------------------------------------------
// fp32 -> bf16 weight conversion (one-shot per launch, deterministic)
// ---------------------------------------------------------------------------
__global__ void cvt_f32_bf16(const float* __restrict__ in,
                             __bf16* __restrict__ out, int n) {
  int i = blockIdx.x * blockDim.x + threadIdx.x;
  int stride = gridDim.x * blockDim.x;
  for (; i < n; i += stride) out[i] = (__bf16)in[i];
}

// ---------------------------------------------------------------------------
// reductions
// ---------------------------------------------------------------------------
__device__ __forceinline__ float wave_sum(float v) {
#pragma unroll
  for (int o = 16; o > 0; o >>= 1) v += __shfl_xor(v, o, 32);
  return v;
}

// ---------------------------------------------------------------------------
// Kernel 1: LayerNorm1 + cyclic shift (-SHIFT) + cast to bf16
// out[b, i, :] = LN(x[b, (i+SHIFT) mod N, :])   (shifted coordinates)
// one block (128 threads) per token row; 4 f32 per thread (512 = 128*4)
// ---------------------------------------------------------------------------
__global__ __launch_bounds__(128)
void ln_shift_kernel(const float* __restrict__ x,
                     const float* __restrict__ w,
                     const float* __restrict__ b,
                     __bf16* __restrict__ out) {
  int t  = blockIdx.x;
  int bi = t / FRAMES, n = t % FRAMES;
  int ns = n + SHIFT; if (ns >= FRAMES) ns -= FRAMES;
  const float4* row = (const float4*)(x + ((size_t)bi * FRAMES + ns) * DIM);
  int tid = threadIdx.x;
  float4 xv = row[tid];
  float s  = xv.x + xv.y + xv.z + xv.w;
  float sq = xv.x*xv.x + xv.y*xv.y + xv.z*xv.z + xv.w*xv.w;
  __shared__ float rs[4], rq[4];
  int lane = tid & 31, wid = tid >> 5;
  s = wave_sum(s); sq = wave_sum(sq);
  if (lane == 0) { rs[wid] = s; rq[wid] = sq; }
  __syncthreads();
  s  = rs[0] + rs[1] + rs[2] + rs[3];
  sq = rq[0] + rq[1] + rq[2] + rq[3];
  float mean = s * (1.0f / DIM);
  float var  = sq * (1.0f / DIM) - mean * mean;
  float rstd = rsqrtf(var + LN_EPS);
  int c = tid * 4;
  __bf16* o = out + (size_t)t * DIM + c;
  o[0] = (__bf16)((xv.x - mean) * rstd * w[c+0] + b[c+0]);
  o[1] = (__bf16)((xv.y - mean) * rstd * w[c+1] + b[c+1]);
  o[2] = (__bf16)((xv.z - mean) * rstd * w[c+2] + b[c+2]);
  o[3] = (__bf16)((xv.w - mean) * rstd * w[c+3] + b[c+3]);
}

// ---------------------------------------------------------------------------
// Kernel 2: residual + reverse shift + LayerNorm2
// x1[b,n] = x[b,n] + proj[b, (n-SHIFT) mod N]          (f32, kept for final res)
// h2[b,n] = LN2(x1[b,n])                               (bf16, feeds FC1)
// ---------------------------------------------------------------------------
__global__ __launch_bounds__(128)
void resid_ln2_kernel(const float* __restrict__ x,
                      const float* __restrict__ proj,   // shifted coords
                      const float* __restrict__ w,
                      const float* __restrict__ b,
                      float* __restrict__ x1,
                      __bf16* __restrict__ h2) {
  int t  = blockIdx.x;
  int bi = t / FRAMES, n = t % FRAMES;
  int ns = n - SHIFT; if (ns < 0) ns += FRAMES;
  const float4* xr = (const float4*)(x + (size_t)t * DIM);
  const float4* pr = (const float4*)(proj + ((size_t)bi * FRAMES + ns) * DIM);
  int tid = threadIdx.x;
  float4 xv = xr[tid], pv = pr[tid];
  float4 r;
  r.x = xv.x + pv.x; r.y = xv.y + pv.y; r.z = xv.z + pv.z; r.w = xv.w + pv.w;
  ((float4*)(x1 + (size_t)t * DIM))[tid] = r;
  float s  = r.x + r.y + r.z + r.w;
  float sq = r.x*r.x + r.y*r.y + r.z*r.z + r.w*r.w;
  __shared__ float rs[4], rq[4];
  int lane = tid & 31, wid = tid >> 5;
  s = wave_sum(s); sq = wave_sum(sq);
  if (lane == 0) { rs[wid] = s; rq[wid] = sq; }
  __syncthreads();
  s  = rs[0] + rs[1] + rs[2] + rs[3];
  sq = rq[0] + rq[1] + rq[2] + rq[3];
  float mean = s * (1.0f / DIM);
  float var  = sq * (1.0f / DIM) - mean * mean;
  float rstd = rsqrtf(var + LN_EPS);
  int c = tid * 4;
  __bf16* o = h2 + (size_t)t * DIM + c;
  o[0] = (__bf16)((r.x - mean) * rstd * w[c+0] + b[c+0]);
  o[1] = (__bf16)((r.y - mean) * rstd * w[c+1] + b[c+1]);
  o[2] = (__bf16)((r.z - mean) * rstd * w[c+2] + b[c+2]);
  o[3] = (__bf16)((r.w - mean) * rstd * w[c+3] + b[c+3]);
}

// ---------------------------------------------------------------------------
// Kernel 3: windowed attention. One block (128 threads) per (window, head).
// qkv rows are in shifted coordinates: window w covers tokens [10w, 10w+10).
// Only the last window of each 1000-frame sequence carries the shift mask.
// ---------------------------------------------------------------------------
__global__ __launch_bounds__(128)
void attn_kernel(const __bf16* __restrict__ qkv,       // (TOKENS, 1536)
                 const float* __restrict__ bias_tab,   // (2*WIN-1, HEADS)
                 __bf16* __restrict__ out) {           // (TOKENS, DIM)
  __shared__ float qs[WIN][HD];
  __shared__ float ks[WIN][HD];
  __shared__ float vs[WIN][HD];
  __shared__ float sc[WIN][WIN];
  int blk = blockIdx.x;
  int h   = blk & 7;
  int w   = blk >> 3;                  // 0 .. 6399
  int wl  = w % (FRAMES / WIN);        // window index within sequence
  size_t t0 = (size_t)w * WIN;
  int tid = threadIdx.x;

  for (int idx = tid; idx < WIN * HD; idx += 128) {
    int r = idx >> 6, d = idx & 63;
    size_t base = (t0 + r) * (3 * DIM) + (size_t)h * HD + d;
    qs[r][d] = (float)qkv[base]            * 0.125f;   // hd^-0.5 = 1/8
    ks[r][d] = (float)qkv[base + DIM];
    vs[r][d] = (float)qkv[base + 2 * DIM];
  }
  __syncthreads();

  if (tid < WIN * WIN) {
    int qi = tid / WIN, kj = tid % WIN;
    float s = 0.f;
#pragma unroll
    for (int d = 0; d < HD; ++d) s += qs[qi][d] * ks[kj][d];
    s += bias_tab[(qi - kj + WIN - 1) * HEADS + h];
    if (wl == (FRAMES / WIN - 1) && ((qi < SHIFT) != (kj < SHIFT))) s -= 100.0f;
    sc[qi][kj] = s;
  }
  __syncthreads();

  if (tid < WIN) {
    float m = sc[tid][0];
#pragma unroll
    for (int j = 1; j < WIN; ++j) m = fmaxf(m, sc[tid][j]);
    float e[WIN]; float sum = 0.f;
#pragma unroll
    for (int j = 0; j < WIN; ++j) { e[j] = expf(sc[tid][j] - m); sum += e[j]; }
    float inv = 1.0f / sum;
#pragma unroll
    for (int j = 0; j < WIN; ++j) sc[tid][j] = e[j] * inv;
  }
  __syncthreads();

  for (int idx = tid; idx < WIN * HD; idx += 128) {
    int r = idx >> 6, d = idx & 63;
    float o = 0.f;
#pragma unroll
    for (int k = 0; k < WIN; ++k) o += sc[r][k] * vs[k][d];
    out[(t0 + r) * DIM + (size_t)h * HD + d] = (__bf16)o;
  }
}

// ---------------------------------------------------------------------------
// Kernel 4: bf16 GEMM via V_WMMA_F32_16X16X32_BF16, f32 accumulate.
//   D[M,N] = A[M,K] * W[N,K]^T + bias[N]   (torch Linear layout)
// Block tile 128x128, K-step 32, 8 waves (2x4), 4x2 WMMA tiles per wave,
// double-buffered LDS, padded to kill bank conflicts.
// EPI: 0 = bf16 store, 1 = bf16 GELU(exact) store, 2 = f32 store,
//      3 = f32 store + residual add (res[M,N])
// ---------------------------------------------------------------------------
template<int EPI>
__global__ __launch_bounds__(256)
void gemm_wmma_bf16(const __bf16* __restrict__ A,
                    const __bf16* __restrict__ W,
                    const float*  __restrict__ bias,
                    const float*  __restrict__ res,
                    __bf16* __restrict__ outb,
                    float*  __restrict__ outf,
                    int M, int N, int K) {
  constexpr int BM = 128, BN = 128, BK = 32, PAD = 40;
  __shared__ __bf16 As[2][BM][PAD];
  __shared__ __bf16 Ws[2][BN][PAD];

  int tid    = threadIdx.x;
  int waveId = tid >> 5;
  int lane   = tid & 31;
  int hi     = lane >> 4;       // lane half (K interleave select)
  int lm     = lane & 15;       // row-within-tile
  int waveM  = waveId >> 2;     // 0..1  -> 64 rows each
  int waveN  = waveId & 3;      // 0..3  -> 32 cols each
  int m0 = blockIdx.y * BM;
  int n0 = blockIdx.x * BN;

  // global->LDS staging map: 512 chunks of 8 bf16, 2 per thread per matrix
  int c0 = tid, c1 = tid + 256;
  int ar0 = c0 >> 2, ac0 = (c0 & 3) << 3;
  int ar1 = c1 >> 2, ac1 = (c1 & 3) << 3;
  const __bf16* Ag0 = A + (size_t)(m0 + ar0) * K + ac0;
  const __bf16* Ag1 = A + (size_t)(m0 + ar1) * K + ac1;
  const __bf16* Wg0 = W + (size_t)(n0 + ar0) * K + ac0;
  const __bf16* Wg1 = W + (size_t)(n0 + ar1) * K + ac1;

  v8f zero = {0.f, 0.f, 0.f, 0.f, 0.f, 0.f, 0.f, 0.f};
  v8f acc[4][2];
#pragma unroll
  for (int i = 0; i < 4; ++i)
#pragma unroll
    for (int j = 0; j < 2; ++j) acc[i][j] = zero;

  int nk = K / BK;

  // prologue: stage K-tile 0
  v8bf a0 = *(const v8bf*)(Ag0);
  v8bf a1 = *(const v8bf*)(Ag1);
  v8bf w0 = *(const v8bf*)(Wg0);
  v8bf w1 = *(const v8bf*)(Wg1);
  *(v8bf*)&As[0][ar0][ac0] = a0;
  *(v8bf*)&As[0][ar1][ac1] = a1;
  *(v8bf*)&Ws[0][ar0][ac0] = w0;
  *(v8bf*)&Ws[0][ar1][ac1] = w1;

  union Frag { v16bf v; v8bf h[2]; };

  for (int kt = 0; kt < nk; ++kt) {
    __syncthreads();
    int cur = kt & 1;

    if (kt + 1 < nk) {                       // issue next-tile global loads early
      size_t ko = (size_t)(kt + 1) * BK;
      a0 = *(const v8bf*)(Ag0 + ko);
      a1 = *(const v8bf*)(Ag1 + ko);
      w0 = *(const v8bf*)(Wg0 + ko);
      w1 = *(const v8bf*)(Wg1 + ko);
      if (kt + 2 < nk) {                     // global_prefetch_b8 two tiles ahead
        __builtin_prefetch(Ag0 + ko + BK, 0, 1);
        __builtin_prefetch(Wg0 + ko + BK, 0, 1);
      }
    }

    // B fragments: lane = output feature; K = {0..7,16..23} / {8..15,24..31} per half
    Frag fb[2];
#pragma unroll
    for (int ni = 0; ni < 2; ++ni) {
      const __bf16* q = &Ws[cur][waveN * 32 + ni * 16 + lm][0];
      fb[ni].h[0] = *(const v8bf*)(q + hi * 8);
      fb[ni].h[1] = *(const v8bf*)(q + 16 + hi * 8);
    }
    // A fragments: lane = M row; K = {0..7,16..23} / {8..15,24..31} per half
#pragma unroll
    for (int mi = 0; mi < 4; ++mi) {
      Frag fa;
      const __bf16* p = &As[cur][waveM * 64 + mi * 16 + lm][0];
      fa.h[0] = *(const v8bf*)(p + hi * 8);
      fa.h[1] = *(const v8bf*)(p + 16 + hi * 8);
#pragma unroll
      for (int ni = 0; ni < 2; ++ni)
        acc[mi][ni] = __builtin_amdgcn_wmma_f32_16x16x32_bf16(
            false, fa.v, false, fb[ni].v, (short)0, acc[mi][ni], false, false);
    }

    if (kt + 1 < nk) {                       // stage next tile into other buffer
      int nxt = cur ^ 1;
      *(v8bf*)&As[nxt][ar0][ac0] = a0;
      *(v8bf*)&As[nxt][ar1][ac1] = a1;
      *(v8bf*)&Ws[nxt][ar0][ac0] = w0;
      *(v8bf*)&Ws[nxt][ar1][ac1] = w1;
    }
  }

  // epilogue: C layout -> lane n = lane&15, row m = vgpr + (lane>>4)*8
  int colBase = n0 + waveN * 32;
#pragma unroll
  for (int ni = 0; ni < 2; ++ni) {
    int col = colBase + ni * 16 + lm;
    float bv = bias[col];
#pragma unroll
    for (int mi = 0; mi < 4; ++mi) {
      int mb = m0 + waveM * 64 + mi * 16 + hi * 8;
#pragma unroll
      for (int r = 0; r < 8; ++r) {
        float v = acc[mi][ni][r] + bv;
        size_t off = (size_t)(mb + r) * N + col;
        if (EPI == 0) {
          outb[off] = (__bf16)v;
        } else if (EPI == 1) {
          v = 0.5f * v * (1.0f + erff(v * 0.70710678118654752f));
          outb[off] = (__bf16)v;
        } else if (EPI == 2) {
          outf[off] = v;
        } else {
          outf[off] = v + res[off];
        }
      }
    }
  }
}

// ---------------------------------------------------------------------------
// launch
// ---------------------------------------------------------------------------
extern "C" void kernel_launch(void* const* d_in, const int* in_sizes, int n_in,
                              void* d_out, int out_size, void* d_ws, size_t ws_size,
                              hipStream_t stream) {
  (void)in_sizes; (void)n_in; (void)out_size; (void)ws_size;

  const float* x      = (const float*)d_in[0];
  const float* n1w    = (const float*)d_in[1];
  const float* n1b    = (const float*)d_in[2];
  const float* qkv_w  = (const float*)d_in[3];
  const float* qkv_b  = (const float*)d_in[4];
  const float* relb   = (const float*)d_in[5];
  const float* proj_w = (const float*)d_in[6];
  const float* proj_b = (const float*)d_in[7];
  const float* n2w    = (const float*)d_in[8];
  const float* n2b    = (const float*)d_in[9];
  const float* fc1_w  = (const float*)d_in[10];
  const float* fc1_b  = (const float*)d_in[11];
  const float* fc2_w  = (const float*)d_in[12];
  const float* fc2_b  = (const float*)d_in[13];

  // workspace regions (aliased across pipeline phases):
  //  region1 (65.5MB):  hs  -> attn_out -> h2
  //  region2 (262MB):   qkv -> proj_f32 -> fc1_out
  //  region3 (131MB):   x1
  //  region4 (6.3MB):   bf16 weights
  char* ws = (char*)d_ws;
  const size_t R1 = (size_t)TOKENS * DIM * 2;        // 65,536,000
  const size_t R2 = (size_t)TOKENS * HIDDEN * 2;     // 262,144,000
  const size_t R3 = (size_t)TOKENS * DIM * 4;        // 131,072,000

  __bf16* hs    = (__bf16*)(ws);
  __bf16* attnO = (__bf16*)(ws);
  __bf16* h2    = (__bf16*)(ws);
  char*   reg2  = ws + R1;
  __bf16* qkv   = (__bf16*)(reg2);
  float*  projf = (float*)(reg2);
  __bf16* fc1o  = (__bf16*)(reg2);
  float*  x1    = (float*)(ws + R1 + R2);
  __bf16* wqkv  = (__bf16*)(ws + R1 + R2 + R3);
  __bf16* wproj = wqkv  + (size_t)(3 * DIM) * DIM;
  __bf16* wfc1  = wproj + (size_t)DIM * DIM;
  __bf16* wfc2  = wfc1  + (size_t)HIDDEN * DIM;

  // 0) weights -> bf16
  cvt_f32_bf16<<<256, 256, 0, stream>>>(qkv_w,  wqkv, 3 * DIM * DIM);
  cvt_f32_bf16<<<256, 256, 0, stream>>>(proj_w, wproj, DIM * DIM);
  cvt_f32_bf16<<<512, 256, 0, stream>>>(fc1_w,  wfc1, HIDDEN * DIM);
  cvt_f32_bf16<<<512, 256, 0, stream>>>(fc2_w,  wfc2, DIM * HIDDEN);

  // 1) LN1 + shift -> bf16
  ln_shift_kernel<<<TOKENS, 128, 0, stream>>>(x, n1w, n1b, hs);

  // 2) QKV GEMM: (64000 x 512) * (1536 x 512)^T -> bf16
  gemm_wmma_bf16<0><<<dim3((3 * DIM) / 128, TOKENS / 128), 256, 0, stream>>>(
      hs, wqkv, qkv_b, nullptr, qkv, nullptr, TOKENS, 3 * DIM, DIM);

  // 3) windowed attention (6400 windows x 8 heads)
  attn_kernel<<<(TOKENS / WIN) * HEADS, 128, 0, stream>>>(qkv, relb, attnO);

  // 4) proj GEMM -> f32 (still shifted coords)
  gemm_wmma_bf16<2><<<dim3(DIM / 128, TOKENS / 128), 256, 0, stream>>>(
      attnO, wproj, proj_b, nullptr, nullptr, projf, TOKENS, DIM, DIM);

  // 5) residual + reverse shift + LN2
  resid_ln2_kernel<<<TOKENS, 128, 0, stream>>>(x, projf, n2w, n2b, x1, h2);

  // 6) FC1 GEMM + exact GELU -> bf16
  gemm_wmma_bf16<1><<<dim3(HIDDEN / 128, TOKENS / 128), 256, 0, stream>>>(
      h2, wfc1, fc1_b, nullptr, fc1o, nullptr, TOKENS, HIDDEN, DIM);

  // 7) FC2 GEMM + residual(x1) -> f32 d_out
  gemm_wmma_bf16<3><<<dim3(DIM / 128, TOKENS / 128), 256, 0, stream>>>(
      fc1o, wfc2, fc2_b, x1, nullptr, (float*)d_out, TOKENS, DIM, HIDDEN);
}
